// TransformerLayer_55259049230489
// MI455X (gfx1250) — matrix operand order repaired and checked by reading, not compiled, not run on hardware
//
#include <hip/hip_runtime.h>

// ---------------------------------------------------------------------------
// CDNA5 (gfx1250) wave32 WMMA transformer layer.
// All matmuls run as bf16 WMMA (v_wmma_f32_16x16x32_bf16) with f32 accum.
// Global->LDS staging uses GLOBAL_LOAD_ASYNC_TO_LDS_B128 (ASYNCcnt) when the
// toolchain exposes the builtin; otherwise falls back to reg staging.
// ---------------------------------------------------------------------------

typedef __attribute__((ext_vector_type(16))) __bf16     v16bf;
typedef __attribute__((ext_vector_type(8)))  float      v8f;
typedef __attribute__((ext_vector_type(4)))  unsigned   v4u;
typedef __attribute__((ext_vector_type(4)))  int        v4i;

union FragU { v4u u2[2]; v16bf bf; };

#if defined(__HIP_DEVICE_COMPILE__) && \
    __has_builtin(__builtin_amdgcn_global_load_async_to_lds_b128) && \
    __has_builtin(__builtin_amdgcn_s_wait_asynccnt)
#define USE_ASYNC_LDS 1
#else
#define USE_ASYNC_LDS 0
#endif

#if USE_ASYNC_LDS
typedef v4i __attribute__((address_space(1)))* glb_v4i_p;
typedef v4i __attribute__((address_space(3)))* lds_v4i_p;

__device__ __forceinline__ glb_v4i_p to_glb(const void* p) {
  // generic global pointer value == AS1 flat address
  return (glb_v4i_p)(unsigned long long)p;
}
__device__ __forceinline__ lds_v4i_p to_lds(const void* p) {
  // generic LDS pointer -> 32-bit LDS offset (matches LLVM addrspacecast)
  return (lds_v4i_p)(unsigned)(unsigned long long)p;
}
#endif

__device__ __forceinline__ unsigned short f2bf(float f) {
  unsigned u = __float_as_uint(f);
  u += 0x7fffu + ((u >> 16) & 1u);          // round-to-nearest-even
  return (unsigned short)(u >> 16);
}

// ---------------------------------------------------------------------------
// elementwise f32 -> bf16 (packed stores)
// ---------------------------------------------------------------------------
__global__ __launch_bounds__(256) void cvt_f32_bf16_k(
    const float* __restrict__ in, unsigned short* __restrict__ out) {
  long i = ((long)blockIdx.x * 256 + threadIdx.x) * 4;
  float4 f = *(const float4*)&in[i];
  unsigned p0 = (unsigned)f2bf(f.x) | ((unsigned)f2bf(f.y) << 16);
  unsigned p1 = (unsigned)f2bf(f.z) | ((unsigned)f2bf(f.w) << 16);
  unsigned* o = (unsigned*)&out[i];
  o[0] = p0; o[1] = p1;
}

// ---------------------------------------------------------------------------
// tiled transpose + f32->bf16:  in [R][C] f32  ->  out [C][R] bf16
// ---------------------------------------------------------------------------
__global__ __launch_bounds__(256) void transpose_cvt_k(
    const float* __restrict__ in, unsigned short* __restrict__ out,
    int R, int C, long inStride, long outStride) {
  __shared__ float tile[32][33];
  const float* ib = in + (long)blockIdx.z * inStride;
  unsigned short* ob = out + (long)blockIdx.z * outStride;
  int c0 = blockIdx.x * 32, r0 = blockIdx.y * 32;
  int tx = threadIdx.x & 31, ty = threadIdx.x >> 5;   // 32 x 8
#pragma unroll
  for (int i = 0; i < 32; i += 8)
    tile[ty + i][tx] = ib[(long)(r0 + ty + i) * C + c0 + tx];
  __syncthreads();
#pragma unroll
  for (int i = 0; i < 32; i += 8)
    ob[(long)(c0 + ty + i) * R + r0 + tx] = f2bf(tile[tx][ty + i]);
}

// ---------------------------------------------------------------------------
// Generic bf16 WMMA GEMM:  C[M,N] = A[M,K] * B'[N,K]^T  (both row-major, K-contig)
// Block tile 128x128x64, 8 waves (4x2), wave tile 32x64 (2x4 WMMA tiles).
// LDS double-buffered: one barrier per K-step; next tile streams in (async
// to LDS when available) while WMMAs run on the current buffer.
// FLAGS: bit0 = bf16 out (pair-packed b32 stores), bit1 = relu,
//        bit2 = has bias, bit3 = has f32 residual
// ---------------------------------------------------------------------------
#define GM_LDK 72   // 64 + 8 pad (bf16 elems) -> 144B row stride, conflict-free

template <int FLAGS>
__global__ __launch_bounds__(256) void gemm_bf16_wmma(
    const unsigned short* __restrict__ A, int lda, long strideA,
    const unsigned short* __restrict__ B, int ldb, long strideB,
    void* __restrict__ C, int ldc, long strideC,
    const float* __restrict__ bias,
    const float* __restrict__ res, int ldres,
    int K) {
  constexpr bool OBF  = (FLAGS & 1) != 0;
  constexpr bool REL  = (FLAGS & 2) != 0;
  constexpr bool HASB = (FLAGS & 4) != 0;
  constexpr bool HASR = (FLAGS & 8) != 0;

  __shared__ __align__(16) unsigned short Ash[2][128 * GM_LDK];
  __shared__ __align__(16) unsigned short Bsh[2][128 * GM_LDK];

  const int tid = threadIdx.x, lane = tid & 31, wave = tid >> 5;
  const int mi = wave >> 1, ni = wave & 1;        // 4 x 2 wave grid
  const int hi = lane >> 4, l15 = lane & 15;
  const int m0 = blockIdx.y * 128, n0 = blockIdx.x * 128;
  const unsigned short* Ab = A + (long)blockIdx.z * strideA;
  const unsigned short* Bb = B + (long)blockIdx.z * strideB;

  const int lrow = tid >> 3;          // +32*i, i = 0..3
  const int lseg = (tid & 7) * 8;

  v8f acc[8] = {};

#if USE_ASYNC_LDS
  auto stage = [&](int buf, int k0) {
#pragma unroll
    for (int i = 0; i < 4; ++i) {
      int row = lrow + 32 * i;
      __builtin_amdgcn_global_load_async_to_lds_b128(
          to_glb(&Ab[(long)(m0 + row) * lda + k0 + lseg]),
          to_lds(&Ash[buf][row * GM_LDK + lseg]), 0, 0);
      __builtin_amdgcn_global_load_async_to_lds_b128(
          to_glb(&Bb[(long)(n0 + row) * ldb + k0 + lseg]),
          to_lds(&Bsh[buf][row * GM_LDK + lseg]), 0, 0);
    }
  };
  stage(0, 0);
  __builtin_amdgcn_s_wait_asynccnt(0);
  __syncthreads();
#else
  v4u ra[4], rbv[4];
  auto gload = [&](int k0) {
#pragma unroll
    for (int i = 0; i < 4; ++i) {
      int row = lrow + 32 * i;
      ra[i]  = *(const v4u*)&Ab[(long)(m0 + row) * lda + k0 + lseg];
      rbv[i] = *(const v4u*)&Bb[(long)(n0 + row) * ldb + k0 + lseg];
    }
  };
  auto lstore = [&](int buf) {
#pragma unroll
    for (int i = 0; i < 4; ++i) {
      int row = lrow + 32 * i;
      *(v4u*)&Ash[buf][row * GM_LDK + lseg] = ra[i];
      *(v4u*)&Bsh[buf][row * GM_LDK + lseg] = rbv[i];
    }
  };
  gload(0);
  lstore(0);
  __syncthreads();
#endif

  int cur = 0;
  for (int k0 = 0; k0 < K; k0 += 64) {
    const bool more = (k0 + 64) < K;
#if USE_ASYNC_LDS
    if (more) stage(cur ^ 1, k0 + 64);       // streams behind the WMMAs below
#else
    if (more) gload(k0 + 64);
#endif
#pragma unroll
    for (int ks = 0; ks < 2; ++ks) {
      FragU a[2], b[4];
#pragma unroll
      for (int rt = 0; rt < 2; ++rt) {
        const unsigned short* p =
            &Ash[cur][(mi * 32 + rt * 16 + l15) * GM_LDK + ks * 32 + hi * 16];
        a[rt].u2[0] = *(const v4u*)p;
        a[rt].u2[1] = *(const v4u*)(p + 8);
      }
#pragma unroll
      for (int ct = 0; ct < 4; ++ct) {
        const unsigned short* p =
            &Bsh[cur][(ni * 64 + ct * 16 + l15) * GM_LDK + ks * 32 + hi * 16];
        b[ct].u2[0] = *(const v4u*)p;
        b[ct].u2[1] = *(const v4u*)(p + 8);
      }
#pragma unroll
      for (int rt = 0; rt < 2; ++rt)
#pragma unroll
        for (int ct = 0; ct < 4; ++ct)
          acc[rt * 4 + ct] = __builtin_amdgcn_wmma_f32_16x16x32_bf16(
              false, a[rt].bf, false, b[ct].bf, (short)0, acc[rt * 4 + ct],
              false, false);
    }
    if (more) {
#if USE_ASYNC_LDS
      __builtin_amdgcn_s_wait_asynccnt(0);
#else
      lstore(cur ^ 1);
#endif
      __syncthreads();
      cur ^= 1;
    }
  }

  float* Cf = (float*)C + (long)blockIdx.z * strideC;
  unsigned short* Ch = (unsigned short*)C + (long)blockIdx.z * strideC;
#pragma unroll
  for (int rt = 0; rt < 2; ++rt) {
#pragma unroll
    for (int ct = 0; ct < 4; ++ct) {
      int n = n0 + ni * 64 + ct * 16 + l15;
      float bv = HASB ? bias[n] : 0.0f;
      v8f v = acc[rt * 4 + ct];
#pragma unroll
      for (int r = 0; r < 8; ++r) {
        long m = m0 + mi * 32 + rt * 16 + r + 8 * hi;
        float val = v[r] + bv;
        if constexpr (HASR) val += res[m * ldres + n];
        if constexpr (REL)  val = fmaxf(val, 0.0f);
        if constexpr (OBF) {
          float pv = __shfl_xor(val, 1);
          if ((lane & 1) == 0) {
            unsigned pk = (unsigned)f2bf(val) | ((unsigned)f2bf(pv) << 16);
            *(unsigned*)&Ch[m * ldc + n] = pk;
          }
        } else {
          Cf[m * ldc + n] = val;
        }
      }
    }
  }
}

// ---------------------------------------------------------------------------
// Flash attention (causal).  Per block: 1 head h, 1 batch b, 32 query rows.
// Q,K: [H][4096][128] bf16 (tokens = batch-concat).  Vt: [H][1024 dv][4096 tok].
// attn out: [4096][8192] bf16 (head-major concat, pair-packed stores).
// Async path: next K block streams into LDS behind softmax + P*V compute.
// ---------------------------------------------------------------------------
#define AT_QLD 136
#define AT_SLD 68
#define AT_PLD 72

__global__ __launch_bounds__(256) void attn_flash_k(
    const unsigned short* __restrict__ Q, const unsigned short* __restrict__ Km,
    const unsigned short* __restrict__ Vt, unsigned short* __restrict__ attn) {
  __shared__ __align__(16) unsigned short Qsh[32 * AT_QLD];
  __shared__ __align__(16) unsigned short Ksh[64 * AT_QLD];
  __shared__ __align__(16) float          Ssh[32 * AT_SLD];
  __shared__ __align__(16) unsigned short Psh[32 * AT_PLD];
  __shared__ float scale_sh[32];

  const int tid = threadIdx.x, lane = tid & 31, wave = tid >> 5;
  const int hi = lane >> 4, l15 = lane & 15;
  const int rb = blockIdx.x, h = blockIdx.y, bb = blockIdx.z;
  const int smi = wave >> 2, sni = wave & 3;     // S tiles: 2x4; PV: rows 16*smi, cols 256*sni
  const int row0 = rb * 32;                      // within batch
  const long tok0 = (long)bb * 2048 + row0;      // global token row base

  const unsigned short* Qh = Q  + (long)h * 4096 * 128;
  const unsigned short* Kh = Km + (long)h * 4096 * 128;
  const unsigned short* Vh = Vt + (long)h * 1024 * 4096;

  const int nblk = (row0 + 31) / 64 + 1;

#if USE_ASYNC_LDS
  auto stageK = [&](int j0) {
#pragma unroll
    for (int i = 0; i < 4; ++i) {
      int idx = tid + 256 * i;
      int row = idx >> 4, seg = (idx & 15) * 8;
      __builtin_amdgcn_global_load_async_to_lds_b128(
          to_glb(&Kh[((long)bb * 2048 + j0 + row) * 128 + seg]),
          to_lds(&Ksh[row * AT_QLD + seg]), 0, 0);
    }
  };
#pragma unroll
  for (int i = 0; i < 2; ++i) {                  // Q tile 32x128 -> LDS (async)
    int idx = tid + 256 * i;
    int row = idx >> 4, seg = (idx & 15) * 8;
    __builtin_amdgcn_global_load_async_to_lds_b128(
        to_glb(&Qh[(tok0 + row) * 128 + seg]),
        to_lds(&Qsh[row * AT_QLD + seg]), 0, 0);
  }
  stageK(0);
  __builtin_amdgcn_s_wait_asynccnt(0);
  __syncthreads();
#else
#pragma unroll
  for (int i = 0; i < 2; ++i) {                  // Q tile 32x128 -> LDS
    int idx = tid + 256 * i;
    int row = idx >> 4, seg = (idx & 15) * 8;
    *(v4u*)&Qsh[row * AT_QLD + seg] = *(const v4u*)&Qh[(tok0 + row) * 128 + seg];
  }
#endif

  v8f acc[16] = {};                              // 16 rows x 256 dv cols per wave
  float m_r = -3.0e38f, l_r = 0.0f;

  for (int bi = 0; bi < nblk; ++bi) {
    const int j0 = bi * 64;
#if !USE_ASYNC_LDS
#pragma unroll
    for (int i = 0; i < 4; ++i) {                // K block 64x128 -> LDS
      int idx = tid + 256 * i;
      int row = idx >> 4, seg = (idx & 15) * 8;
      *(v4u*)&Ksh[row * AT_QLD + seg] =
          *(const v4u*)&Kh[((long)bb * 2048 + j0 + row) * 128 + seg];
    }
    if (bi + 1 < nblk)
      __builtin_prefetch(&Kh[((long)bb * 2048 + j0 + 64 + (tid >> 2)) * 128], 0, 0);
    __syncthreads();
#endif

    {                                            // S tile (16x16) per wave, DK=128 -> 4 wmma
      v8f s = {};
#pragma unroll
      for (int ks = 0; ks < 4; ++ks) {
        FragU a, b;
        const unsigned short* pa = &Qsh[(smi * 16 + l15) * AT_QLD + ks * 32 + hi * 16];
        a.u2[0] = *(const v4u*)pa; a.u2[1] = *(const v4u*)(pa + 8);
        const unsigned short* pb = &Ksh[(sni * 16 + l15) * AT_QLD + ks * 32 + hi * 16];
        b.u2[0] = *(const v4u*)pb; b.u2[1] = *(const v4u*)(pb + 8);
        s = __builtin_amdgcn_wmma_f32_16x16x32_bf16(false, a.bf, false, b.bf,
                                                    (short)0, s, false, false);
      }
#pragma unroll
      for (int r = 0; r < 8; ++r)
        Ssh[(smi * 16 + r + 8 * hi) * AT_SLD + sni * 16 + l15] = s[r];
    }
    __syncthreads();

#if USE_ASYNC_LDS
    if (bi + 1 < nblk) stageK(j0 + 64);          // K fetch hides behind softmax+PV
#endif

    if (tid < 32) {                              // online softmax, one row per lane
      const int r = tid, grow = row0 + r;
      const float qs = 0.08838834764831845f;     // 1/sqrt(128)
      float p[64], bmax = -3.0e38f;
#pragma unroll
      for (int c = 0; c < 64; ++c) {
        float sv = Ssh[r * AT_SLD + c] * qs;
        sv = ((j0 + c) <= grow) ? sv : -3.0e38f;
        p[c] = sv;
        bmax = fmaxf(bmax, sv);
      }
      float mnew = fmaxf(m_r, bmax);
      float sc = __expf(m_r - mnew);
      float sum = 0.0f;
#pragma unroll
      for (int c = 0; c < 64; c += 2) {
        float e0 = (p[c]     > -1.0e38f) ? __expf(p[c]     - mnew) : 0.0f;
        float e1 = (p[c + 1] > -1.0e38f) ? __expf(p[c + 1] - mnew) : 0.0f;
        sum += e0 + e1;
        *(unsigned*)&Psh[r * AT_PLD + c] =
            (unsigned)f2bf(e0) | ((unsigned)f2bf(e1) << 16);
      }
      l_r = l_r * sc + sum;
      m_r = mnew;
      scale_sh[r] = sc;
    }
    __syncthreads();

    float rs[8];                                 // rescale running accumulators
#pragma unroll
    for (int r = 0; r < 8; ++r) rs[r] = scale_sh[smi * 16 + r + 8 * hi];
#pragma unroll
    for (int t = 0; t < 16; ++t)
#pragma unroll
      for (int r = 0; r < 8; ++r) acc[t][r] *= rs[r];

#pragma unroll
    for (int ks = 0; ks < 2; ++ks) {             // P[16x64] x V[64 x 256] per wave
      FragU a;
      const unsigned short* pa = &Psh[(smi * 16 + l15) * AT_PLD + ks * 32 + hi * 16];
      a.u2[0] = *(const v4u*)pa; a.u2[1] = *(const v4u*)(pa + 8);
      const long tokb = (long)bb * 2048 + j0 + ks * 32 + hi * 16;
#pragma unroll
      for (int ct = 0; ct < 16; ++ct) {
        FragU b;
        const unsigned short* pb = &Vh[(long)(sni * 256 + ct * 16 + l15) * 4096 + tokb];
        b.u2[0] = *(const v4u*)pb; b.u2[1] = *(const v4u*)(pb + 8);
        acc[ct] = __builtin_amdgcn_wmma_f32_16x16x32_bf16(false, a.bf, false, b.bf,
                                                          (short)0, acc[ct], false, false);
      }
    }
#if USE_ASYNC_LDS
    if (bi + 1 < nblk) __builtin_amdgcn_s_wait_asynccnt(0);
#endif
    __syncthreads();
  }

  if (tid < 32) scale_sh[tid] = 1.0f / l_r;      // final 1/l
  __syncthreads();
  float il[8];
#pragma unroll
  for (int r = 0; r < 8; ++r) il[r] = scale_sh[smi * 16 + r + 8 * hi];
#pragma unroll
  for (int ct = 0; ct < 16; ++ct) {
    int col = h * 1024 + sni * 256 + ct * 16 + l15;
#pragma unroll
    for (int r = 0; r < 8; ++r) {
      float val = acc[ct][r] * il[r];
      float pv = __shfl_xor(val, 1);
      if ((lane & 1) == 0) {
        long m = tok0 + smi * 16 + r + 8 * hi;
        *(unsigned*)&attn[m * 8192 + col] =
            (unsigned)f2bf(val) | ((unsigned)f2bf(pv) << 16);
      }
    }
  }
}

// ---------------------------------------------------------------------------
// LayerNorm over 1024 features; one row per block; optional bf16 copy out.
// ---------------------------------------------------------------------------
__global__ __launch_bounds__(256) void layernorm_k(
    const float* __restrict__ in, const float* __restrict__ g,
    const float* __restrict__ be, float* __restrict__ outf,
    unsigned short* __restrict__ outh) {
  __shared__ float red[16];
  const long row = blockIdx.x;
  const int tid = threadIdx.x;
  const float* x = in + row * 1024;
  float v[4], s = 0.0f, sq = 0.0f;
#pragma unroll
  for (int i = 0; i < 4; ++i) {
    v[i] = x[tid + 256 * i];
    s += v[i]; sq += v[i] * v[i];
  }
#pragma unroll
  for (int o = 16; o; o >>= 1) { s += __shfl_xor(s, o); sq += __shfl_xor(sq, o); }
  if ((tid & 31) == 0) { red[tid >> 5] = s; red[8 + (tid >> 5)] = sq; }
  __syncthreads();
  float ts = 0.0f, tq = 0.0f;
#pragma unroll
  for (int i = 0; i < 8; ++i) { ts += red[i]; tq += red[8 + i]; }
  const float mu = ts * (1.0f / 1024.0f);
  const float var = tq * (1.0f / 1024.0f) - mu * mu;
  const float rstd = rsqrtf(var + 1e-5f);
#pragma unroll
  for (int i = 0; i < 4; ++i) {
    int c = tid + 256 * i;
    float y = (v[i] - mu) * rstd * g[c] + be[c];
    outf[row * 1024 + c] = y;
    if (outh) outh[row * 1024 + c] = f2bf(y);
  }
}

// ---------------------------------------------------------------------------
// host-side orchestration
// ---------------------------------------------------------------------------
extern "C" void kernel_launch(void* const* d_in, const int* in_sizes, int n_in,
                              void* d_out, int out_size, void* d_ws, size_t ws_size,
                              hipStream_t stream) {
  (void)in_sizes; (void)n_in; (void)out_size; (void)ws_size;
  const float* x    = (const float*)d_in[0];
  const float* Wq   = (const float*)d_in[1];
  const float* Wk   = (const float*)d_in[2];
  const float* Wv   = (const float*)d_in[3];
  const float* Wm   = (const float*)d_in[4];
  const float* bm   = (const float*)d_in[5];
  const float* ln_g = (const float*)d_in[6];
  const float* ln_b = (const float*)d_in[7];
  const float* W1   = (const float*)d_in[8];
  const float* b1   = (const float*)d_in[9];
  const float* W2   = (const float*)d_in[10];
  const float* b2   = (const float*)d_in[11];

  char* ws = (char*)d_ws;
  size_t off = 0;
  auto alloc = [&](size_t bytes) -> void* {
    void* p = ws + off; off += (bytes + 255) & ~(size_t)255; return p;
  };
  unsigned short* xb   = (unsigned short*)alloc((size_t)4096 * 1024 * 2);
  unsigned short* WqT  = (unsigned short*)alloc((size_t)8 * 128 * 1024 * 2);
  unsigned short* WkT  = (unsigned short*)alloc((size_t)8 * 128 * 1024 * 2);
  unsigned short* WvT  = (unsigned short*)alloc((size_t)8 * 1024 * 1024 * 2);
  unsigned short* WmT  = (unsigned short*)alloc((size_t)1024 * 8192 * 2);
  unsigned short* W1T  = (unsigned short*)alloc((size_t)4096 * 1024 * 2);
  unsigned short* W2T  = (unsigned short*)alloc((size_t)1024 * 4096 * 2);
  unsigned short* Qb   = (unsigned short*)alloc((size_t)8 * 4096 * 128 * 2);
  unsigned short* Kb   = (unsigned short*)alloc((size_t)8 * 4096 * 128 * 2);
  unsigned short* Vt   = (unsigned short*)alloc((size_t)8 * 1024 * 4096 * 2);
  unsigned short* attn = (unsigned short*)alloc((size_t)4096 * 8192 * 2);
  float*          mrg  = (float*)alloc((size_t)4096 * 1024 * 4);
  float*          y1f  = (float*)alloc((size_t)4096 * 1024 * 4);
  unsigned short* y1b  = (unsigned short*)alloc((size_t)4096 * 1024 * 2);
  unsigned short* h1   = (unsigned short*)alloc((size_t)4096 * 4096 * 2);
  float*          ff   = (float*)alloc((size_t)4096 * 1024 * 4);

  // preprocessing: bf16 conversion + weight transposes
  cvt_f32_bf16_k<<<4096, 256, 0, stream>>>(x, xb);
  transpose_cvt_k<<<dim3(4, 32, 8),   256, 0, stream>>>(Wq, WqT, 1024, 128,  131072, 131072);
  transpose_cvt_k<<<dim3(4, 32, 8),   256, 0, stream>>>(Wk, WkT, 1024, 128,  131072, 131072);
  transpose_cvt_k<<<dim3(32, 32, 8),  256, 0, stream>>>(Wv, WvT, 1024, 1024, 1048576, 1048576);
  transpose_cvt_k<<<dim3(32, 256, 1), 256, 0, stream>>>(Wm, WmT, 8192, 1024, 0, 0);
  transpose_cvt_k<<<dim3(128, 32, 1), 256, 0, stream>>>(W1, W1T, 1024, 4096, 0, 0);
  transpose_cvt_k<<<dim3(32, 128, 1), 256, 0, stream>>>(W2, W2T, 4096, 1024, 0, 0);

  // projections: Q, K [H][4096][128] bf16;  Vt = (x Wv)^T -> [H][1024 dv][4096 tok]
  gemm_bf16_wmma<1><<<dim3(1, 32, 8), 256, 0, stream>>>(
      xb, 1024, 0, WqT, 1024, 131072, Qb, 128, 524288, nullptr, nullptr, 0, 1024);
  gemm_bf16_wmma<1><<<dim3(1, 32, 8), 256, 0, stream>>>(
      xb, 1024, 0, WkT, 1024, 131072, Kb, 128, 524288, nullptr, nullptr, 0, 1024);
  gemm_bf16_wmma<1><<<dim3(32, 8, 8), 256, 0, stream>>>(
      WvT, 1024, 1048576, xb, 1024, 0, Vt, 4096, 4194304, nullptr, nullptr, 0, 1024);

  // causal flash attention -> attn [4096][8192] bf16
  attn_flash_k<<<dim3(64, 8, 2), 256, 0, stream>>>(Qb, Kb, Vt, attn);

  // merge: attn @ Wm + bm + x  (f32)
  gemm_bf16_wmma<12><<<dim3(8, 32, 1), 256, 0, stream>>>(
      attn, 8192, 0, WmT, 8192, 0, mrg, 1024, 0, bm, x, 1024, 8192);
  layernorm_k<<<4096, 256, 0, stream>>>(mrg, ln_g, ln_b, y1f, y1b);

  // FFN
  gemm_bf16_wmma<7><<<dim3(32, 32, 1), 256, 0, stream>>>(
      y1b, 1024, 0, W1T, 1024, 0, h1, 4096, 0, b1, nullptr, 0, 1024);
  gemm_bf16_wmma<12><<<dim3(8, 32, 1), 256, 0, stream>>>(
      h1, 4096, 0, W2T, 4096, 0, ff, 1024, 0, b2, y1f, 1024, 4096);
  layernorm_k<<<4096, 256, 0, stream>>>(ff, ln_g, ln_b, (float*)d_out, nullptr);
}
